// EncoderBlock_9972914061605
// MI455X (gfx1250) — matrix operand order repaired and checked by reading, not compiled, not run on hardware
//
#include <hip/hip_runtime.h>

typedef __bf16 v16bf __attribute__((ext_vector_type(16)));
typedef float  v8f   __attribute__((ext_vector_type(8)));
typedef unsigned short u16;

union ABFrag { v16bf v; uint4 u[2]; };

__device__ __forceinline__ u16 f2bf(float f) {
  unsigned int u = __float_as_uint(f);
  u += 0x7FFFu + ((u >> 16) & 1u);   // round-to-nearest-even
  return (u16)(u >> 16);
}

__device__ __forceinline__ v8f wmma_bf16(v16bf a, v16bf b, v8f c) {
  // D = A(16x32 bf16) * B(32x16 bf16) + C(16x16 f32)
  return __builtin_amdgcn_wmma_f32_16x16x32_bf16(false, a, false, b, (short)0, c, false, false);
}

// Cross-lane helpers via ds_swizzle_b32 (group-of-32 mode, immediate masks).
template <int MASK>
__device__ __forceinline__ float swz_xor(float x) {
  return __int_as_float(__builtin_amdgcn_ds_swizzle(__float_as_int(x), (MASK << 10) | 31));
}
__device__ __forceinline__ float swz_bcast16(float x) {
  // src lane = (lane & 16): lane0 -> lanes 0..15, lane16 -> lanes 16..31
  return __int_as_float(__builtin_amdgcn_ds_swizzle(__float_as_int(x), 16));
}
__device__ __forceinline__ float wave_red_sum(float x) {
  x += swz_xor<1>(x); x += swz_xor<2>(x); x += swz_xor<4>(x);
  x += swz_xor<8>(x); x += swz_xor<16>(x);
  return x;
}

// gfx1250 async global->LDS copy (tracked by ASYNCcnt), per CDNA5 ISA ch.10.
__device__ __forceinline__ void async_load_b128(unsigned lds_off,
                                                unsigned long long base,
                                                unsigned voff) {
  asm volatile("global_load_async_to_lds_b128 %0, %1, %2"
               :: "v"(lds_off), "v"(voff), "s"(base) : "memory");
}
__device__ __forceinline__ void wait_asynccnt0() {
  asm volatile("s_wait_asynccnt 0x0" ::: "memory");
}

// ---------------------------------------------------------------------------
// Generic bf16-A x fp32-weight GEMM with fused epilogues.
//   C[M,N] = A[M,K](bf16) @ W[K,N](fp32 -> bf16 during LDS staging)
// 128x128 block tile, K-step 32, 8 waves (4m x 2n), wave tile 32x64.
// A panel: async global->LDS (double-buffered LDS, ASYNCcnt).
// W panel: register-staged double buffering with on-the-fly fp32->bf16.
// ---------------------------------------------------------------------------
constexpr int EPI_QKV   = 0;
constexpr int EPI_OPROJ = 1;
constexpr int EPI_MOE1  = 2;
constexpr int EPI_MOE2  = 3;

constexpr int BM = 128, BN = 128, BK = 32, LDS_S = BK + 8; // 40 elems, 80B rows

template <int EPI>
__global__ __launch_bounds__(256) void gemm_bf16(
    const u16* __restrict__ A, const float* __restrict__ W,
    int M, int N, int K,
    const float* __restrict__ bias, float scale,
    const float* __restrict__ resid, const float* __restrict__ wgt,
    int e, int accflag,
    u16* __restrict__ out_bf, float* __restrict__ out_f) {
  __shared__ __align__(16) u16 As[2][BM][LDS_S];
  __shared__ __align__(16) u16 Bs[BN][LDS_S];

  const int tid  = threadIdx.x;
  const int lane = tid & 31, wid = tid >> 5;
  const int hl = lane >> 4, ln = lane & 15;
  const int wm = wid >> 1, wn = wid & 1;
  const int bm0 = blockIdx.y * BM, bn0 = blockIdx.x * BN;

  // per-thread A staging coordinates (two 16B chunks per thread)
  const int arow0 = tid >> 2,          ach0 = (tid & 3) * 8;
  const int arow1 = (tid + 256) >> 2,  ach1 = ((tid + 256) & 3) * 8;
  v8f acc[2][4] = {};
  float4 wr[4];

  const unsigned long long abase = (unsigned long long)(uintptr_t)A;
  const unsigned lds0 = (unsigned)(uintptr_t)&As[0][arow0][ach0];
  const unsigned lds1 = (unsigned)(uintptr_t)&As[0][arow1][ach1];
  const unsigned bufstride = (unsigned)sizeof(As[0]);

  auto issueA = [&](int k0, int buf) {
    unsigned v0 = (unsigned)(((size_t)(bm0 + arow0) * K + k0 + ach0) * 2);
    unsigned v1 = (unsigned)(((size_t)(bm0 + arow1) * K + k0 + ach1) * 2);
    async_load_b128(lds0 + buf * bufstride, abase, v0);
    async_load_b128(lds1 + buf * bufstride, abase, v1);
  };
  auto loadW = [&](int k0) {
    #pragma unroll
    for (int p = 0; p < 4; ++p) {
      int idx = tid + p * 256;
      int kr = idx >> 5, c4 = (idx & 31) * 4;
      wr[p] = *reinterpret_cast<const float4*>(W + (size_t)(k0 + kr) * N + bn0 + c4);
    }
  };

  issueA(0, 0);
  loadW(0);
  int buf = 0;
  for (int k0 = 0; k0 < K; k0 += BK, buf ^= 1) {
    // commit staged W registers to LDS (fp32 -> bf16)
    #pragma unroll
    for (int p = 0; p < 4; ++p) {
      int idx = tid + p * 256;
      int kr = idx >> 5, c4 = (idx & 31) * 4;
      Bs[c4 + 0][kr] = f2bf(wr[p].x);
      Bs[c4 + 1][kr] = f2bf(wr[p].y);
      Bs[c4 + 2][kr] = f2bf(wr[p].z);
      Bs[c4 + 3][kr] = f2bf(wr[p].w);
    }
    wait_asynccnt0();            // my async A loads into As[buf] are done
    __syncthreads();             // everyone's A + W panels visible
    if (k0 + BK < K) {           // overlap next tile with this tile's WMMAs
      issueA(k0 + BK, buf ^ 1);
      loadW(k0 + BK);
      __builtin_prefetch(W + (size_t)(k0 + 2 * BK) * N + bn0, 0, 0);
    }

    ABFrag af[2], bf[4];
    #pragma unroll
    for (int sm = 0; sm < 2; ++sm) {
      int r = wm * 32 + sm * 16 + ln, kb = hl * 8;
      af[sm].u[0] = *reinterpret_cast<const uint4*>(&As[buf][r][kb]);
      af[sm].u[1] = *reinterpret_cast<const uint4*>(&As[buf][r][kb + 16]);
    }
    #pragma unroll
    for (int sn = 0; sn < 4; ++sn) {
      int c = wn * 64 + sn * 16 + ln, kb = hl * 8;
      bf[sn].u[0] = *reinterpret_cast<const uint4*>(&Bs[c][kb]);
      bf[sn].u[1] = *reinterpret_cast<const uint4*>(&Bs[c][kb + 16]);
    }
    #pragma unroll
    for (int sm = 0; sm < 2; ++sm)
      #pragma unroll
      for (int sn = 0; sn < 4; ++sn)
        acc[sm][sn] = wmma_bf16(af[sm].v, bf[sn].v, acc[sm][sn]);
    __syncthreads();
  }

  // ---- epilogue ----
  #pragma unroll
  for (int sm = 0; sm < 2; ++sm) {
    #pragma unroll
    for (int sn = 0; sn < 4; ++sn) {
      int col  = bn0 + wn * 64 + sn * 16 + ln;
      int rowb = bm0 + wm * 32 + sm * 16 + hl * 8;
      #pragma unroll
      for (int v = 0; v < 8; ++v) {
        int row = rowb + v;
        size_t oi = (size_t)row * N + col;
        float a = acc[sm][sn][v];
        if (EPI == EPI_QKV) {
          out_bf[oi] = f2bf((a + bias[col]) * scale);
        } else if (EPI == EPI_OPROJ) {
          out_f[oi] = a + bias[col] + resid[oi];
        } else if (EPI == EPI_MOE1) {
          float t = a + bias[col];
          t = t > 0.f ? t : 0.f;
          out_bf[oi] = f2bf(t * wgt[row * 8 + e]);
        } else { // EPI_MOE2
          float t = a + wgt[row * 8 + e] * bias[col];
          if (accflag) t += out_f[oi];
          out_f[oi] = t;
        }
      }
    }
  }
}

// ---------------------------------------------------------------------------
// Flash attention: grid(x = S/128, y = B*H). 8 waves x 16 query rows.
// l accumulated via a WMMA "ones column" (o[4]); row max via batched swizzles.
// ---------------------------------------------------------------------------
__global__ __launch_bounds__(256) void attn_kernel(
    const u16* __restrict__ qb, const u16* __restrict__ kbuf,
    const u16* __restrict__ vbuf, u16* __restrict__ ob, int S) {
  __shared__ __align__(16) u16 Ks[32][72];     // [key][dh]
  __shared__ __align__(16) u16 Vt[64][40];     // [dh][key]
  __shared__ __align__(16) u16 Ps[8][16][40];  // per-wave P tile

  const int tid = threadIdx.x, lane = tid & 31, wid = tid >> 5;
  const int hl = lane >> 4, ln = lane & 15;
  const int bh = blockIdx.y;
  const int b = bh >> 4, h = bh & 15;
  const int q0 = blockIdx.x * 128 + wid * 16;

  ABFrag qa[2];
  {
    int m = q0 + ln;
    const u16* qp = qb + ((size_t)(b * S + m) * 16 + h) * 64;
    int kb = hl * 8;
    qa[0].u[0] = *reinterpret_cast<const uint4*>(qp + kb);
    qa[0].u[1] = *reinterpret_cast<const uint4*>(qp + kb + 16);
    qa[1].u[0] = *reinterpret_cast<const uint4*>(qp + 32 + kb);
    qa[1].u[1] = *reinterpret_cast<const uint4*>(qp + 48 + kb);
  }
  // "ones column" B fragment: column n==0 is 1.0, rest 0 -> row sums
  ABFrag onef;
  {
    unsigned hb = (ln == 0) ? 0x3F80u : 0u;
    unsigned wrd = hb | (hb << 16);
    onef.u[0] = make_uint4(wrd, wrd, wrd, wrd);
    onef.u[1] = make_uint4(wrd, wrd, wrd, wrd);
  }

  float mrow[8];
  v8f o[5] = {};   // o[0..3] = output DH tiles, o[4] = l
  #pragma unroll
  for (int v = 0; v < 8; ++v) mrow[v] = -1e30f;

  const int key = tid >> 3, c = (tid & 7) * 8;
  uint4 kreg, vreg;
  auto ldchunk = [&](int kt) {
    size_t base = ((size_t)(b * S + kt + key) * 16 + h) * 64 + c;
    kreg = *reinterpret_cast<const uint4*>(kbuf + base);
    vreg = *reinterpret_cast<const uint4*>(vbuf + base);
  };
  ldchunk(0);

  for (int kt = 0; kt < S; kt += 32) {
    *reinterpret_cast<uint4*>(&Ks[key][c]) = kreg;
    {
      u16 tmp[8];
      *reinterpret_cast<uint4*>(tmp) = vreg;
      #pragma unroll
      for (int j = 0; j < 8; ++j) Vt[c + j][key] = tmp[j];
    }
    __syncthreads();
    if (kt + 32 < S) ldchunk(kt + 32);

    // S = Q K^T for 32 keys (two 16-key subtiles x two DH halves)
    v8f s[2];
    #pragma unroll
    for (int nt = 0; nt < 2; ++nt) {
      int keyc = nt * 16 + ln, kb2 = hl * 8;
      ABFrag kf0, kf1;
      kf0.u[0] = *reinterpret_cast<const uint4*>(&Ks[keyc][kb2]);
      kf0.u[1] = *reinterpret_cast<const uint4*>(&Ks[keyc][kb2 + 16]);
      kf1.u[0] = *reinterpret_cast<const uint4*>(&Ks[keyc][32 + kb2]);
      kf1.u[1] = *reinterpret_cast<const uint4*>(&Ks[keyc][32 + kb2 + 16]);
      v8f sv = {};
      sv = wmma_bf16(qa[0].v, kf0.v, sv);
      sv = wmma_bf16(qa[1].v, kf1.v, sv);
      s[nt] = sv;
    }

    // batched row-max reduction: issue all 8 swizzles per step back-to-back
    float t[8];
    #pragma unroll
    for (int v = 0; v < 8; ++v) t[v] = fmaxf(s[0][v], s[1][v]);
    {
      float tmp[8];
      #pragma unroll
      for (int v = 0; v < 8; ++v) tmp[v] = swz_xor<1>(t[v]);
      #pragma unroll
      for (int v = 0; v < 8; ++v) t[v] = fmaxf(t[v], tmp[v]);
      #pragma unroll
      for (int v = 0; v < 8; ++v) tmp[v] = swz_xor<2>(t[v]);
      #pragma unroll
      for (int v = 0; v < 8; ++v) t[v] = fmaxf(t[v], tmp[v]);
      #pragma unroll
      for (int v = 0; v < 8; ++v) tmp[v] = swz_xor<4>(t[v]);
      #pragma unroll
      for (int v = 0; v < 8; ++v) t[v] = fmaxf(t[v], tmp[v]);
      #pragma unroll
      for (int v = 0; v < 8; ++v) tmp[v] = swz_xor<8>(t[v]);
      #pragma unroll
      for (int v = 0; v < 8; ++v) t[v] = fmaxf(t[v], tmp[v]);
    }

    // online softmax update; P -> LDS for A-fragment re-read
    #pragma unroll
    for (int v = 0; v < 8; ++v) {
      float mnew  = fmaxf(mrow[v], t[v]);
      float alpha = __expf(mrow[v] - mnew);
      float p0 = __expf(s[0][v] - mnew);
      float p1 = __expf(s[1][v] - mnew);
      mrow[v] = mnew;
      #pragma unroll
      for (int t5 = 0; t5 < 5; ++t5) o[t5][v] *= alpha;  // rescale O and l
      int pr = v + 8 * hl;
      Ps[wid][pr][ln]      = f2bf(p0);
      Ps[wid][pr][16 + ln] = f2bf(p1);
    }

    // O += P(16x32) . V(32x64), plus l += P . ones
    ABFrag pf;
    pf.u[0] = *reinterpret_cast<const uint4*>(&Ps[wid][ln][hl * 8]);
    pf.u[1] = *reinterpret_cast<const uint4*>(&Ps[wid][ln][hl * 8 + 16]);
    #pragma unroll
    for (int dt = 0; dt < 4; ++dt) {
      int dh = dt * 16 + ln;
      ABFrag vf;
      vf.u[0] = *reinterpret_cast<const uint4*>(&Vt[dh][hl * 8]);
      vf.u[1] = *reinterpret_cast<const uint4*>(&Vt[dh][hl * 8 + 16]);
      o[dt] = wmma_bf16(pf.v, vf.v, o[dt]);
    }
    o[4] = wmma_bf16(pf.v, onef.v, o[4]);
    __syncthreads();
  }

  #pragma unroll
  for (int v = 0; v < 8; ++v) {
    float inv = 1.0f / swz_bcast16(o[4][v]);
    int row = q0 + v + 8 * hl;
    #pragma unroll
    for (int dt = 0; dt < 4; ++dt)
      ob[((size_t)(b * S + row) * 16 + h) * 64 + dt * 16 + ln] =
          f2bf(o[dt][v] * inv);
  }
}

// ---------------------------------------------------------------------------
// LayerNorm (one wave per token, D=1024). Optional residual add, dual output.
// ---------------------------------------------------------------------------
__global__ __launch_bounds__(256) void ln_kernel(
    const float* __restrict__ in, const float* __restrict__ resid,
    const float* __restrict__ g, const float* __restrict__ be,
    float* __restrict__ out_f, u16* __restrict__ out_bf) {
  const int lane = threadIdx.x & 31, wid = threadIdx.x >> 5;
  const int t = blockIdx.x * 8 + wid;
  const float* p = in + (size_t)t * 1024;
  float vals[32];
  float s = 0.f;
  #pragma unroll
  for (int i = 0; i < 32; ++i) {
    float v = p[lane + i * 32];
    if (resid) v += resid[(size_t)t * 1024 + lane + i * 32];
    vals[i] = v;
    s += v;
  }
  s = wave_red_sum(s);
  float mu = s * (1.0f / 1024.0f);
  float vs = 0.f;
  #pragma unroll
  for (int i = 0; i < 32; ++i) { float d = vals[i] - mu; vs += d * d; }
  vs = wave_red_sum(vs);
  float rstd = rsqrtf(vs * (1.0f / 1024.0f) + 1e-5f);
  #pragma unroll
  for (int i = 0; i < 32; ++i) {
    int d = lane + i * 32;
    float y = (vals[i] - mu) * rstd * g[d] + be[d];
    if (out_f)  out_f[(size_t)t * 1024 + d] = y;
    if (out_bf) out_bf[(size_t)t * 1024 + d] = f2bf(y);
  }
}

// ---------------------------------------------------------------------------
// Gate: logits[m,g,e] = sum_d x[m,d]*gate_w[g,d,e]; per-gate softmax over e,
// averaged over G=2 -> w[m,e]. One wave per token.
// ---------------------------------------------------------------------------
__global__ __launch_bounds__(256) void gate_kernel(
    const float* __restrict__ xn, const float* __restrict__ gw,
    float* __restrict__ wout) {
  const int lane = threadIdx.x & 31, wid = threadIdx.x >> 5;
  const int t = blockIdx.x * 8 + wid;
  float acc[16] = {};
  const float* xp = xn + (size_t)t * 1024;
  for (int d = lane; d < 1024; d += 32) {
    float xv = xp[d];
    #pragma unroll
    for (int e = 0; e < 8; ++e) acc[e] += xv * gw[d * 8 + e];
    #pragma unroll
    for (int e = 0; e < 8; ++e) acc[8 + e] += xv * gw[8192 + d * 8 + e];
  }
  #pragma unroll
  for (int i = 0; i < 16; ++i) acc[i] = wave_red_sum(acc[i]);

  float w[8];
  #pragma unroll
  for (int e = 0; e < 8; ++e) w[e] = 0.f;
  #pragma unroll
  for (int g = 0; g < 2; ++g) {
    float mx = acc[g * 8];
    #pragma unroll
    for (int e = 1; e < 8; ++e) mx = fmaxf(mx, acc[g * 8 + e]);
    float pe[8], sum = 0.f;
    #pragma unroll
    for (int e = 0; e < 8; ++e) { pe[e] = __expf(acc[g * 8 + e] - mx); sum += pe[e]; }
    float inv = 0.5f / sum;  // averaged over G=2
    #pragma unroll
    for (int e = 0; e < 8; ++e) w[e] += pe[e] * inv;
  }
  if (lane == 0) {
    #pragma unroll
    for (int e = 0; e < 8; ++e) wout[t * 8 + e] = w[e];
  }
}

__global__ __launch_bounds__(256) void cvt_bf16_kernel(
    const float* __restrict__ in, u16* __restrict__ out) {
  int i = blockIdx.x * 256 + threadIdx.x;
  out[i] = f2bf(in[i]);
}

// ---------------------------------------------------------------------------
extern "C" void kernel_launch(void* const* d_in, const int* in_sizes, int n_in,
                              void* d_out, int out_size, void* d_ws, size_t ws_size,
                              hipStream_t stream) {
  (void)in_sizes; (void)n_in; (void)out_size; (void)ws_size;
  constexpr int Bb = 2, S = 1024, D = 1024, V = 4096, E = 8;
  constexpr int M = Bb * S;

  const float* x     = (const float*)d_in[0];
  const float* wq    = (const float*)d_in[1];
  const float* bq    = (const float*)d_in[2];
  const float* wk    = (const float*)d_in[3];
  const float* bk    = (const float*)d_in[4];
  const float* wv    = (const float*)d_in[5];
  const float* bv    = (const float*)d_in[6];
  const float* wo    = (const float*)d_in[7];
  const float* bo    = (const float*)d_in[8];
  const float* ln1g  = (const float*)d_in[9];
  const float* ln1b  = (const float*)d_in[10];
  const float* ln2g  = (const float*)d_in[11];
  const float* ln2b  = (const float*)d_in[12];
  const float* gatew = (const float*)d_in[13];
  const float* ew1   = (const float*)d_in[14];
  const float* eb1   = (const float*)d_in[15];
  const float* ew2   = (const float*)d_in[16];
  const float* eb2   = (const float*)d_in[17];

  char* ws = (char*)d_ws;
  size_t off = 0;
  auto alloc = [&](size_t bytes) {
    off = (off + 255) & ~(size_t)255;
    size_t o = off; off += bytes; return o;
  };
  u16*   xb    = (u16*)(ws + alloc((size_t)M * D * 2));
  u16*   qb    = (u16*)(ws + alloc((size_t)M * D * 2));
  u16*   kb    = (u16*)(ws + alloc((size_t)M * D * 2));
  u16*   vb    = (u16*)(ws + alloc((size_t)M * D * 2));
  u16*   ab    = (u16*)(ws + alloc((size_t)M * D * 2));
  float* res1  = (float*)(ws + alloc((size_t)M * D * 4));
  float* nrm1  = (float*)(ws + alloc((size_t)M * D * 4));
  u16*   nrm1b = (u16*)(ws + alloc((size_t)M * D * 2));
  float* wgt   = (float*)(ws + alloc((size_t)M * E * 4));
  u16*   he    = (u16*)(ws + alloc((size_t)M * V * 2));
  float* moe   = (float*)(ws + alloc((size_t)M * D * 4));

  const float qscale = 0.125f;  // 1/sqrt(DH)

  cvt_bf16_kernel<<<(M * D) / 256, 256, 0, stream>>>(x, xb);

  dim3 blk(256);
  dim3 gProj(D / BN, M / BM);   // (8,16)
  gemm_bf16<EPI_QKV><<<gProj, blk, 0, stream>>>(xb, wq, M, D, D, bq, qscale,
                                                nullptr, nullptr, 0, 0, qb, nullptr);
  gemm_bf16<EPI_QKV><<<gProj, blk, 0, stream>>>(xb, wk, M, D, D, bk, 1.0f,
                                                nullptr, nullptr, 0, 0, kb, nullptr);
  gemm_bf16<EPI_QKV><<<gProj, blk, 0, stream>>>(xb, wv, M, D, D, bv, 1.0f,
                                                nullptr, nullptr, 0, 0, vb, nullptr);
  attn_kernel<<<dim3(S / 128, Bb * 16), blk, 0, stream>>>(qb, kb, vb, ab, S);
  gemm_bf16<EPI_OPROJ><<<gProj, blk, 0, stream>>>(ab, wo, M, D, D, bo, 1.0f,
                                                  x, nullptr, 0, 0, nullptr, res1);
  ln_kernel<<<M / 8, blk, 0, stream>>>(res1, nullptr, ln1g, ln1b, nrm1, nrm1b);
  gate_kernel<<<M / 8, blk, 0, stream>>>(nrm1, gatew, wgt);
  for (int e = 0; e < E; ++e) {
    gemm_bf16<EPI_MOE1><<<dim3(V / BN, M / BM), blk, 0, stream>>>(
        nrm1b, ew1 + (size_t)e * D * V, M, V, D, eb1 + (size_t)e * V, 1.0f,
        nullptr, wgt, e, 0, he, nullptr);
    gemm_bf16<EPI_MOE2><<<gProj, blk, 0, stream>>>(
        he, ew2 + (size_t)e * V * D, M, D, V, eb2 + (size_t)e * D, 1.0f,
        nullptr, wgt, e, (e > 0) ? 1 : 0, nullptr, moe);
  }
  ln_kernel<<<M / 8, blk, 0, stream>>>(moe, nrm1, ln2g, ln2b, (float*)d_out, nullptr);
}
